// GCN_19748259627189
// MI455X (gfx1250) — matrix operand order repaired and checked by reading, not compiled, not run on hardware
//
#include <hip/hip_runtime.h>

// ---------------------------------------------------------------------------
// 2-layer GCN (PyG GCNConv w/ self loops + edge weights) + softmax for
// MI455X / gfx1250.
//  - GEMMs: bf16 WMMA (v_wmma_f32_16x16x32_bf16), fp32 accumulate.
//  - W-matrix slabs double-buffered in LDS via gfx1250 async-to-LDS DMA
//    (global_load_async_to_lds_b128 + s_wait_asynccnt).
//  - Aggregation: L2-resident global_atomic_add_f32 scatter.
// ---------------------------------------------------------------------------

typedef __attribute__((ext_vector_type(16))) __bf16 v16bf;
typedef __attribute__((ext_vector_type(8)))  float  v8f;

__device__ __forceinline__ unsigned lds_off32(const void* p) {
  // generic pointer to LDS: addrspacecast(3->0) = {aperture_hi, lds_offset}
  return (unsigned)(unsigned long long)p;
}

__device__ __forceinline__ void async_copy_b128(unsigned lds_byte_off,
                                                const float* gsrc) {
  asm volatile("global_load_async_to_lds_b128 %0, %1, off"
               :: "v"(lds_byte_off), "v"(gsrc) : "memory");
}

// ------------------------------ degree / dinv ------------------------------

__global__ void k_init_deg(float* __restrict__ deg, int N) {
  int i = blockIdx.x * blockDim.x + threadIdx.x;
  if (i < N) deg[i] = 1.0f;                 // self-loop weight = 1
}

__global__ void k_deg_scatter(const int* __restrict__ col,
                              const float* __restrict__ ew,
                              float* __restrict__ deg, int E) {
  int e = blockIdx.x * blockDim.x + threadIdx.x;
  if (e < E) atomicAdd(&deg[col[e]], ew[e]);
}

__global__ void k_dinv(float* __restrict__ deg, int N) {
  int i = blockIdx.x * blockDim.x + threadIdx.x;
  if (i < N) {
    float d = deg[i];
    deg[i] = (d > 0.0f) ? rsqrtf(d) : 0.0f;
  }
}

// ------------------------------ GEMM1 (WMMA) -------------------------------
// h1[N x 64] = x[N x 512] @ W1[512 x 64]
// Block: 128 threads = 4 waves; block tile = 32 rows x 64 cols.
// Wave w -> column tile w; each wave holds two 16x16 accumulators (32 rows).
// W1 slabs (32x64 fp32 = 8KB) double-buffered in LDS via async DMA.

__global__ __launch_bounds__(128) void k_gemm1_wmma(
    const float* __restrict__ x, const float* __restrict__ W1,
    float* __restrict__ h1, int N, int K, int F /*64*/) {
  __shared__ __align__(16) float shW[2][32 * 64];   // 2 x 8KB

  const int lane    = threadIdx.x & 31;
  const int wave    = threadIdx.x >> 5;            // 0..3 -> column tile
  const int rowBase = blockIdx.x * 32;
  const int colBase = wave * 16;
  const int half    = lane >> 4;                   // 0: lanes 0-15, 1: 16-31
  const int l15     = lane & 15;

  int r0 = rowBase + l15;       if (r0 >= N) r0 = N - 1;   // clamp loads only
  int r1 = rowBase + 16 + l15;  if (r1 >= N) r1 = N - 1;
  const float* __restrict__ xr0 = x + (size_t)r0 * K;
  const float* __restrict__ xr1 = x + (size_t)r1 * K;

  // prologue: stage W1 slab 0 into shW[0] (4 x b128 per thread = 8KB/block)
  {
    unsigned lbase = lds_off32(&shW[0][0]) + threadIdx.x * 16u;
    const float* gp = W1 + (size_t)threadIdx.x * 4;
#pragma unroll
    for (int rr = 0; rr < 4; ++rr)
      async_copy_b128(lbase + rr * 2048u, gp + (size_t)rr * 512);
  }

  v8f acc0 = {}, acc1 = {};
  const int T = K / 32;
  for (int i = 0; i < T; ++i) {
    const int buf = i & 1;
    if (i + 1 < T) {   // prefetch slab i+1 into the other buffer
      unsigned lbase = lds_off32(&shW[buf ^ 1][0]) + threadIdx.x * 16u;
      const float* gp = W1 + (size_t)(i + 1) * 32 * F + (size_t)threadIdx.x * 4;
#pragma unroll
      for (int rr = 0; rr < 4; ++rr)
        async_copy_b128(lbase + rr * 2048u, gp + (size_t)rr * 512);
      asm volatile("s_wait_asynccnt 0x4" ::: "memory");  // slab i landed
    } else {
      asm volatile("s_wait_asynccnt 0x0" ::: "memory");
    }
    __syncthreads();                                     // publish slab i

    const int k0 = i * 32;
    if (k0 + 32 < K) __builtin_prefetch(xr0 + k0 + 32, 0, 1);

    // A fragments (16x32 bf16): lanes<16 K={0..7,16..23}, lanes>=16 +8
    const int kb = k0 + half * 8;
    float4 p0 = *(const float4*)(xr0 + kb);
    float4 p1 = *(const float4*)(xr0 + kb + 4);
    float4 p2 = *(const float4*)(xr0 + kb + 16);
    float4 p3 = *(const float4*)(xr0 + kb + 20);
    float4 q0 = *(const float4*)(xr1 + kb);
    float4 q1 = *(const float4*)(xr1 + kb + 4);
    float4 q2 = *(const float4*)(xr1 + kb + 16);
    float4 q3 = *(const float4*)(xr1 + kb + 20);
    v16bf a0, a1;
    a0[0]=(__bf16)p0.x; a0[1]=(__bf16)p0.y; a0[2]=(__bf16)p0.z; a0[3]=(__bf16)p0.w;
    a0[4]=(__bf16)p1.x; a0[5]=(__bf16)p1.y; a0[6]=(__bf16)p1.z; a0[7]=(__bf16)p1.w;
    a0[8]=(__bf16)p2.x; a0[9]=(__bf16)p2.y; a0[10]=(__bf16)p2.z; a0[11]=(__bf16)p2.w;
    a0[12]=(__bf16)p3.x; a0[13]=(__bf16)p3.y; a0[14]=(__bf16)p3.z; a0[15]=(__bf16)p3.w;
    a1[0]=(__bf16)q0.x; a1[1]=(__bf16)q0.y; a1[2]=(__bf16)q0.z; a1[3]=(__bf16)q0.w;
    a1[4]=(__bf16)q1.x; a1[5]=(__bf16)q1.y; a1[6]=(__bf16)q1.z; a1[7]=(__bf16)q1.w;
    a1[8]=(__bf16)q2.x; a1[9]=(__bf16)q2.y; a1[10]=(__bf16)q2.z; a1[11]=(__bf16)q2.w;
    a1[12]=(__bf16)q3.x; a1[13]=(__bf16)q3.y; a1[14]=(__bf16)q3.z; a1[15]=(__bf16)q3.w;

    // B fragment (32x16 bf16) from LDS: lane -> column, half -> K block
    v16bf bfr;
#pragma unroll
    for (int j = 0; j < 16; ++j)
      bfr[j] = (__bf16)shW[buf][(half * 8 + j + (j >= 8 ? 8 : 0)) * 64 + colBase + l15];

    acc0 = __builtin_amdgcn_wmma_f32_16x16x32_bf16(
        false, a0, false, bfr, (short)0, acc0, false, false);
    acc1 = __builtin_amdgcn_wmma_f32_16x16x32_bf16(
        false, a1, false, bfr, (short)0, acc1, false, false);

    __syncthreads();   // close WAR window before DMA reuses this buffer
  }

  // C layout: acc[j] -> (M = half*8 + j, N = l15)
  const int ccol = colBase + l15;
  if (rowBase + 32 <= N) {         // uniform fast path (N % 32 == 0 here)
#pragma unroll
    for (int j = 0; j < 8; ++j) {
      h1[(size_t)(rowBase + half * 8 + j) * F + ccol]      = acc0[j];
      h1[(size_t)(rowBase + 16 + half * 8 + j) * F + ccol] = acc1[j];
    }
  } else {
#pragma unroll
    for (int j = 0; j < 8; ++j) {
      int ra = rowBase + half * 8 + j, rb = ra + 16;
      if (ra < N) h1[(size_t)ra * F + ccol] = acc0[j];
      if (rb < N) h1[(size_t)rb * F + ccol] = acc1[j];
    }
  }
}

// ------------------------------ GEMM2 (WMMA) -------------------------------
// h2[N x Fout] = (agg1[N x 64] + b1) @ W2[64 x Fout], Fout=20 padded to 32.

template <int K>
__global__ __launch_bounds__(64) void k_gemm2_wmma(
    const float* __restrict__ agg1, const float* __restrict__ b1,
    const float* __restrict__ W2, float* __restrict__ h2,
    int N, int Fout /*20*/) {
  const int lane    = threadIdx.x & 31;
  const int wave    = threadIdx.x >> 5;          // 0..1 -> column tile
  const int rowBase = blockIdx.x * 16;
  const int colBase = wave * 16;
  const int half    = lane >> 4;
  const int l15     = lane & 15;
  const int col     = colBase + l15;

  int arow = rowBase + l15;
  if (arow >= N) arow = N - 1;
  const float* __restrict__ hrow = agg1 + (size_t)arow * K;

  v8f acc = {};
#pragma unroll
  for (int k0 = 0; k0 < K; k0 += 32) {
    const int kb = k0 + half * 8;
    float4 p0 = *(const float4*)(hrow + kb);
    float4 p1 = *(const float4*)(hrow + kb + 4);
    float4 p2 = *(const float4*)(hrow + kb + 16);
    float4 p3 = *(const float4*)(hrow + kb + 20);
    v16bf a;
    a[0]=(__bf16)(p0.x + b1[kb+0]);  a[1]=(__bf16)(p0.y + b1[kb+1]);
    a[2]=(__bf16)(p0.z + b1[kb+2]);  a[3]=(__bf16)(p0.w + b1[kb+3]);
    a[4]=(__bf16)(p1.x + b1[kb+4]);  a[5]=(__bf16)(p1.y + b1[kb+5]);
    a[6]=(__bf16)(p1.z + b1[kb+6]);  a[7]=(__bf16)(p1.w + b1[kb+7]);
    a[8]=(__bf16)(p2.x + b1[kb+16]); a[9]=(__bf16)(p2.y + b1[kb+17]);
    a[10]=(__bf16)(p2.z + b1[kb+18]); a[11]=(__bf16)(p2.w + b1[kb+19]);
    a[12]=(__bf16)(p3.x + b1[kb+20]); a[13]=(__bf16)(p3.y + b1[kb+21]);
    a[14]=(__bf16)(p3.z + b1[kb+22]); a[15]=(__bf16)(p3.w + b1[kb+23]);

    v16bf b;
    const int krow = k0 + half * 8;
#pragma unroll
    for (int j = 0; j < 16; ++j)
      b[j] = (col < Fout) ? (__bf16)W2[(size_t)(krow + j + (j >= 8 ? 8 : 0)) * Fout + col]
                          : (__bf16)0.0f;

    acc = __builtin_amdgcn_wmma_f32_16x16x32_bf16(
        false, a, false, b, (short)0, acc, false, false);
  }

#pragma unroll
  for (int j = 0; j < 8; ++j) {
    int r = rowBase + half * 8 + j;
    if (r < N && col < Fout) h2[(size_t)r * Fout + col] = acc[j];
  }
}

// --------------------------- aggregation kernels ---------------------------

// agg[v][f] = dinv[v]^2 * h[v][f]   (self-loop contribution, also zero-inits)
template <int F>
__global__ void k_agg_init(const float* __restrict__ h,
                           const float* __restrict__ dinv,
                           float* __restrict__ agg, int N, int Fvalid) {
  long long t = (long long)blockIdx.x * blockDim.x + threadIdx.x;
  int v = (int)(t / F), f = (int)(t % F);
  if (v >= N || f >= Fvalid) return;
  float di = dinv[v];
  agg[(size_t)v * Fvalid + f] = di * di * h[(size_t)v * Fvalid + f];
}

// agg[col] += dinv[row]*ew*dinv[col] * h[row]   (F threads per edge)
template <int F>
__global__ __launch_bounds__(256) void k_edge_scatter(
    const int* __restrict__ row, const int* __restrict__ col,
    const float* __restrict__ ew, const float* __restrict__ dinv,
    const float* __restrict__ h, float* __restrict__ agg,
    int E, int Fvalid) {
  long long t = (long long)blockIdx.x * blockDim.x + threadIdx.x;
  int e = (int)(t / F), f = (int)(t % F);
  if (e >= E || f >= Fvalid) return;
  int r = row[e], c = col[e];
  float norm = dinv[r] * ew[e] * dinv[c];
  atomicAdd(&agg[(size_t)c * Fvalid + f], norm * h[(size_t)r * Fvalid + f]);
}

// ------------------------------ softmax + bias -----------------------------

__global__ __launch_bounds__(256) void k_softmax(float* __restrict__ out,
                                                 const float* __restrict__ b2,
                                                 int N, int Fout) {
  int wave = threadIdx.x >> 5;
  int lane = threadIdx.x & 31;
  int node = blockIdx.x * 8 + wave;
  if (node >= N) return;
  float* __restrict__ p = out + (size_t)node * Fout;

  float v = -3.402823466e38f;
  if (lane < Fout) v = p[lane] + b2[lane];
  float m = v;
#pragma unroll
  for (int off = 16; off > 0; off >>= 1) m = fmaxf(m, __shfl_xor(m, off, 32));
  float e = (lane < Fout) ? __expf(v - m) : 0.0f;
  float s = e;
#pragma unroll
  for (int off = 16; off > 0; off >>= 1) s += __shfl_xor(s, off, 32);
  if (lane < Fout) p[lane] = e / s;
}

// --------------------------------- launch ----------------------------------

static inline size_t align256(size_t x) { return (x + 255) & ~(size_t)255; }

extern "C" void kernel_launch(void* const* d_in, const int* in_sizes, int n_in,
                              void* d_out, int out_size, void* d_ws, size_t ws_size,
                              hipStream_t stream) {
  const float* x    = (const float*)d_in[0];
  const float* ew   = (const float*)d_in[1];
  const float* W1   = (const float*)d_in[2];
  const float* b1   = (const float*)d_in[3];
  const float* W2   = (const float*)d_in[4];
  const float* b2   = (const float*)d_in[5];
  const int*   eidx = (const int*)  d_in[6];

  const int E    = in_sizes[1];
  const int Fhid = in_sizes[3];                 // 64
  const int Fout = in_sizes[5];                 // 20
  const int Fin  = in_sizes[2] / Fhid;          // 512
  const int N    = in_sizes[0] / Fin;           // 100000
  const int* row = eidx;
  const int* col = eidx + E;

  // workspace: dinv | h1 (later reused as h2) | agg1
  char*  ws   = (char*)d_ws;
  float* dinv = (float*)ws;
  size_t off  = align256((size_t)N * sizeof(float));
  float* h1   = (float*)(ws + off);
  off        += align256((size_t)N * Fhid * sizeof(float));
  float* agg1 = (float*)(ws + off);
  float* h2   = h1;                             // h1 dead after scatter1
  float* outp = (float*)d_out;

  const int B = 256;

  // 1) deg -> dinv (self-loop folded into init value 1.0)
  k_init_deg<<<(N + B - 1) / B, B, 0, stream>>>(dinv, N);
  k_deg_scatter<<<(E + B - 1) / B, B, 0, stream>>>(col, ew, dinv, E);
  k_dinv<<<(N + B - 1) / B, B, 0, stream>>>(dinv, N);

  // 2) h1 = x @ W1  (bf16 WMMA, async-LDS double-buffered W slabs)
  k_gemm1_wmma<<<(N + 31) / 32, 128, 0, stream>>>(x, W1, h1, N, Fin, Fhid);

  // 3) agg1 = dinv^2*h1 (self loop) + edge scatter
  {
    long long tot = (long long)N * 64;
    k_agg_init<64><<<(int)((tot + B - 1) / B), B, 0, stream>>>(h1, dinv, agg1, N, Fhid);
    tot = (long long)E * 64;
    k_edge_scatter<64><<<(int)((tot + B - 1) / B), B, 0, stream>>>(
        row, col, ew, dinv, h1, agg1, E, Fhid);
  }

  // 4) h2 = (agg1 + b1) @ W2  (WMMA, 2 col tiles of 16, cols >= Fout masked)
  k_gemm2_wmma<64><<<(N + 15) / 16, 32 * ((Fout + 15) / 16), 0, stream>>>(
      agg1, b1, W2, h2, N, Fout);

  // 5) out = dinv^2*h2 (self loop) + edge scatter, straight into d_out
  {
    long long tot = (long long)N * 32;
    k_agg_init<32><<<(int)((tot + B - 1) / B), B, 0, stream>>>(h2, dinv, outp, N, Fout);
    tot = (long long)E * 32;
    k_edge_scatter<32><<<(int)((tot + B - 1) / B), B, 0, stream>>>(
        row, col, ew, dinv, h2, outp, E, Fout);
  }

  // 6) softmax(out + b2) in place, one wave per node
  k_softmax<<<(N + 7) / 8, B, 0, stream>>>(outp, b2, N, Fout);
}
